// LocalFeatureAggregation_12936441495845
// MI455X (gfx1250) — matrix operand order repaired and checked by reading, not compiled, not run on hardware
//
#include <hip/hip_runtime.h>
#include <hip/hip_bf16.h>

typedef __attribute__((ext_vector_type(16))) __bf16 v16bf;
typedef __attribute__((ext_vector_type(8)))  __bf16 v8bf;
typedef __attribute__((ext_vector_type(8)))  float  v8f;

#define C_IN  128
#define C2    256
#define C_OUT 128
#define KNBR  16

// ---------------------------------------------------------------------------
// WMMA helpers (CDNA5, wave32): D = A(16x32 bf16) * B(32x16 bf16) + C(f32)
// ---------------------------------------------------------------------------
__device__ __forceinline__ v8f wmma_bf16(v16bf a, v16bf b, v8f c) {
    return __builtin_amdgcn_wmma_f32_16x16x32_bf16(
        /*neg_a=*/false, a, /*neg_b=*/false, b,
        /*c_mod=*/(short)0, c, /*reuse_a=*/false, /*reuse_b=*/false);
}

// A-operand (16x32, MxK) from LDS, row-major [16][ldk] bf16.
// Contiguous k order, as in B: lane L -> row M=L%16; a[j]=K[k0+16*half+j], j=0..15
__device__ __forceinline__ v16bf load_a_lds(const __bf16* g, int ldk, int k0, int lane) {
    int m = lane & 15, half = lane >> 4;
    const __bf16* p = g + m * ldk + k0 + 16 * half;
    v8bf lo = *(const v8bf*)p;
    v8bf hi = *(const v8bf*)(p + 8);
    v16bf a;
#pragma unroll
    for (int j = 0; j < 8; ++j) { a[j] = lo[j]; a[8 + j] = hi[j]; }
    return a;
}

// B-operand (32x16, KxN) from a pre-transposed global weight Wt[n][k] (row stride ldk).
// Contiguous k order, as in A: lane L -> col N=ncol+L%16; b[j]=K[k0+16*half+j], j=0..15 (32 contiguous bytes)
__device__ __forceinline__ v16bf load_b_glb(const __bf16* __restrict__ Wt, int ldk,
                                            int ncol, int k0, int lane) {
    int nl = lane & 15, half = lane >> 4;
    return *(const v16bf*)(Wt + (size_t)(ncol + nl) * ldk + k0 + 16 * half);
}

// ---------------------------------------------------------------------------
// Setup: transpose fp32 weight [R][C] -> bf16 [C][R] so B loads are contiguous
// ---------------------------------------------------------------------------
__global__ void transpose_bf16(const float* __restrict__ src, __bf16* __restrict__ dst,
                               int R, int C) {
    int i = blockIdx.x * blockDim.x + threadIdx.x;
    if (i >= R * C) return;
    int r = i / C, c = i - r * C;
    dst[(size_t)c * R + r] = (__bf16)src[i];
}

// ---------------------------------------------------------------------------
// Kernel 1: y = BN(relu(features @ W1 + b1))   (BN folded: gather commutes)
// grid = N/16 blocks x 128 threads (4 waves, each wave 4 N-tiles)
// ---------------------------------------------------------------------------
__global__ __launch_bounds__(128) void k1_mlp_bn(
    const float* __restrict__ features, const __bf16* __restrict__ W1t,
    const float* __restrict__ b1, const float* __restrict__ gamma,
    const float* __restrict__ beta, const float* __restrict__ rmean,
    const float* __restrict__ rvar, float* __restrict__ y) {
    __shared__ __align__(32) __bf16 At[16 * C_IN];
    int tid = threadIdx.x;
    int row0 = blockIdx.x * 16;

    // 16x128 fp32 tile is contiguous in global; convert to bf16 in LDS
    const float* src = features + (size_t)row0 * C_IN;
#pragma unroll
    for (int i = 0; i < 4; ++i) {
        int f = (tid + 128 * i) * 4;
        float4 v = *(const float4*)(src + f);
        At[f + 0] = (__bf16)v.x; At[f + 1] = (__bf16)v.y;
        At[f + 2] = (__bf16)v.z; At[f + 3] = (__bf16)v.w;
    }
    __syncthreads();

    int lane = tid & 31, wid = tid >> 5;
    v8f acc[4] = {};
#pragma unroll
    for (int ks = 0; ks < 4; ++ks) {
        v16bf a = load_a_lds(At, C_IN, ks * 32, lane);
#pragma unroll
        for (int t = 0; t < 4; ++t) {
            v16bf b = load_b_glb(W1t, C_IN, (wid * 4 + t) * 16, ks * 32, lane);
            acc[t] = wmma_bf16(a, b, acc[t]);
        }
    }

    // epilogue: +bias, relu, BN affine.  D layout: VGPR r, lane L -> (M=r+8*(L/16), N=L%16)
    int nl = lane & 15, half = lane >> 4;
#pragma unroll
    for (int t = 0; t < 4; ++t) {
        int c = (wid * 4 + t) * 16 + nl;
        float bb = b1[c];
        float sc = gamma[c] * rsqrtf(rvar[c] + 1e-5f);
        float sh = beta[c] - rmean[c] * sc;
#pragma unroll
        for (int r = 0; r < 8; ++r) {
            int row = row0 + r + 8 * half;
            float v = fmaxf(acc[t][r] + bb, 0.0f);
            y[(size_t)row * C2 + c] = v * sc + sh;
        }
    }
}

// ---------------------------------------------------------------------------
// Kernel 2 (dominant): per node -- gather 16 neighbor rows of y, score GEMM
// (16x256)@(256x256) via WMMA, per-channel softmax over k, weighted sum.
// grid = N blocks x 128 threads (4 waves, each wave 4 N-tiles = 64 channels)
// ---------------------------------------------------------------------------
__global__ __launch_bounds__(128) void k2_attn(
    const float* __restrict__ y, const int* __restrict__ nidx,
    const __bf16* __restrict__ Wst, __bf16* __restrict__ feat) {
    __shared__ __align__(32) float  gf[16 * C2];   // gathered rows, fp32 (16 KB)
    __shared__ __align__(32) __bf16 gb[16 * C2];   // same, bf16 A-operand (8 KB)
    __shared__ __align__(32) float  sb[16 * C2];   // score tile (16 KB)
    int tid = threadIdx.x;
    int node = blockIdx.x;

    // gather: thread t loads 32 channels of neighbor k = t/8
    {
        int k  = tid >> 3;
        int c0 = (tid & 7) * 32;
        int row = nidx[node * KNBR + k];
        const float* srow = y + (size_t)row * C2 + c0;
#pragma unroll
        for (int i = 0; i < 8; ++i) {
            float4 v = *(const float4*)(srow + i * 4);
            int c = k * C2 + c0 + i * 4;
            gf[c + 0] = v.x; gf[c + 1] = v.y; gf[c + 2] = v.z; gf[c + 3] = v.w;
            gb[c + 0] = (__bf16)v.x; gb[c + 1] = (__bf16)v.y;
            gb[c + 2] = (__bf16)v.z; gb[c + 3] = (__bf16)v.w;
        }
    }
    __syncthreads();

    int lane = tid & 31, wid = tid >> 5;
    v8f acc[4] = {};
#pragma unroll
    for (int ks = 0; ks < 8; ++ks) {
        v16bf a = load_a_lds(gb, C2, ks * 32, lane);
#pragma unroll
        for (int t = 0; t < 4; ++t) {
            v16bf b = load_b_glb(Wst, C2, (wid * 4 + t) * 16, ks * 32, lane);
            acc[t] = wmma_bf16(a, b, acc[t]);
        }
    }

    // scores -> LDS (within-wave producer/consumer; DS ops are in-order)
    int nl = lane & 15, half = lane >> 4;
#pragma unroll
    for (int t = 0; t < 4; ++t)
#pragma unroll
        for (int r = 0; r < 8; ++r)
            sb[(r + 8 * half) * C2 + (wid * 4 + t) * 16 + nl] = acc[t][r];

    // per-channel softmax over k=16 neighbors + weighted sum (2 channels/lane)
#pragma unroll
    for (int cc = 0; cc < 2; ++cc) {
        int c = wid * 64 + cc * 32 + lane;
        float mx = -3.0e38f;
#pragma unroll
        for (int k = 0; k < 16; ++k) mx = fmaxf(mx, sb[k * C2 + c]);
        float den = 0.0f, num = 0.0f;
#pragma unroll
        for (int k = 0; k < 16; ++k) {
            float e = __expf(sb[k * C2 + c] - mx);
            den += e;
            num += e * gf[k * C2 + c];
        }
        feat[(size_t)node * C2 + c] = (__bf16)(num / den);
    }
}

// ---------------------------------------------------------------------------
// Kernel 3: out = feat @ Wm + bm.  grid = N/16 x 128 (4 waves, 2 N-tiles each)
// ---------------------------------------------------------------------------
__global__ __launch_bounds__(128) void k3_proj(
    const __bf16* __restrict__ feat, const __bf16* __restrict__ Wmt,
    const float* __restrict__ bm, float* __restrict__ out) {
    __shared__ __align__(32) __bf16 At[16 * C2];   // 8 KB
    int tid = threadIdx.x;
    int row0 = blockIdx.x * 16;

    // 16x256 bf16 tile is one contiguous 8 KB block in global
    const uint4* src = (const uint4*)(feat + (size_t)row0 * C2);
    uint4* dst = (uint4*)At;
#pragma unroll
    for (int i = 0; i < 4; ++i) dst[tid + 128 * i] = src[tid + 128 * i];
    __syncthreads();

    int lane = tid & 31, wid = tid >> 5;
    v8f acc[2] = {};
#pragma unroll
    for (int ks = 0; ks < 8; ++ks) {
        v16bf a = load_a_lds(At, C2, ks * 32, lane);
#pragma unroll
        for (int t = 0; t < 2; ++t) {
            v16bf b = load_b_glb(Wmt, C2, (wid * 2 + t) * 16, ks * 32, lane);
            acc[t] = wmma_bf16(a, b, acc[t]);
        }
    }

    int nl = lane & 15, half = lane >> 4;
#pragma unroll
    for (int t = 0; t < 2; ++t) {
        int c = (wid * 2 + t) * 16 + nl;
        float bb = bm[c];
#pragma unroll
        for (int r = 0; r < 8; ++r)
            out[(size_t)(row0 + r + 8 * half) * C_OUT + c] = acc[t][r] + bb;
    }
}

// ---------------------------------------------------------------------------
extern "C" void kernel_launch(void* const* d_in, const int* in_sizes, int n_in,
                              void* d_out, int out_size, void* d_ws, size_t ws_size,
                              hipStream_t stream) {
    const float* features = (const float*)d_in[0];
    const int*   nidx     = (const int*)d_in[1];
    const float* W1       = (const float*)d_in[2];
    const float* b1       = (const float*)d_in[3];
    const float* gamma    = (const float*)d_in[4];
    const float* beta     = (const float*)d_in[5];
    const float* rmean    = (const float*)d_in[6];
    const float* rvar     = (const float*)d_in[7];
    const float* Ws       = (const float*)d_in[8];
    const float* Wm       = (const float*)d_in[9];
    const float* bm       = (const float*)d_in[10];
    float* out = (float*)d_out;

    const int N = in_sizes[0] / C_IN;   // 50000 (multiple of 16)

    // workspace layout (all offsets 32B-aligned)
    float*  y    = (float*)d_ws;                        // N*256 f32
    __bf16* feat = (__bf16*)(y + (size_t)N * C2);       // N*256 bf16
    __bf16* W1t  = (__bf16*)(feat + (size_t)N * C2);    // [256][128]
    __bf16* Wst  = W1t + C2 * C_IN;                     // [256][256]
    __bf16* Wmt  = Wst + C2 * C2;                       // [128][256]

    transpose_bf16<<<(C_IN * C2 + 255) / 256, 256, 0, stream>>>(W1, W1t, C_IN, C2);
    transpose_bf16<<<(C2 * C2 + 255) / 256, 256, 0, stream>>>(Ws, Wst, C2, C2);
    transpose_bf16<<<(C2 * C_OUT + 255) / 256, 256, 0, stream>>>(Wm, Wmt, C2, C_OUT);

    k1_mlp_bn<<<N / 16, 128, 0, stream>>>(features, W1t, b1, gamma, beta, rmean, rvar, y);
    k2_attn  <<<N,      128, 0, stream>>>(y, nidx, Wst, feat);
    k3_proj  <<<N / 16, 128, 0, stream>>>(feat, Wmt, bm, out);
}
